// MPNN_25220047962166
// MI455X (gfx1250) — compile-verified, matrix operand order checked by reading, not run on hardware
//
#include <hip/hip_runtime.h>
#include <math.h>

#define NATOMS 20000
#define NEDGES 1000000
#define NWAVE  8
#define NSPH   9
#define NORBIT 24
#define DENS_LD 32            // padded leading dim for dens (cols 24..31 stay zero)
#define CUTOFFF 5.0f
#define PI_F 3.14159265358979323846f

typedef __attribute__((ext_vector_type(16))) _Float16 v16h;
typedef __attribute__((ext_vector_type(8)))  float    v8f;

static __device__ __forceinline__ float atomic_add_f32(float* p, float v) {
    return __hip_atomic_fetch_add(p, v, __ATOMIC_RELAXED, __HIP_MEMORY_SCOPE_AGENT);
}

// ---------------------------------------------------------------- zero fill
__global__ void zero_kernel(float* p, int n) {
    int t = blockIdx.x * blockDim.x + threadIdx.x;
    for (int i = t; i < n; i += gridDim.x * blockDim.x) p[i] = 0.0f;
}

// -------------------------------------------------- embedding first layer 1->64
__global__ void emb_l1_kernel(const float* __restrict__ species,
                              const float* __restrict__ w,
                              const float* __restrict__ b,
                              float* __restrict__ h1) {
    int t = blockIdx.x * blockDim.x + threadIdx.x;
    if (t >= NATOMS * 64) return;
    int a = t >> 6, j = t & 63;
    float x = species[a] * w[j] + b[j];
    h1[t] = x / (1.0f + __expf(-x));   // silu
}

// ---------------------------------------------------------------- weight pre-pack
// Pack W[Kreal][Nreal] (f32 row-major) into WMMA B-fragment lane layout:
// Bpk[(kt*nNT + nt)*32 + lane][16] f16, 32B contiguous per lane.
__global__ void pack_w_kernel(const float* __restrict__ W, int Kreal, int Nreal,
                              int nKT, int nNT, _Float16* __restrict__ Bpk) {
    int t = blockIdx.x * blockDim.x + threadIdx.x;
    if (t >= nKT * nNT * 32) return;
    int lane = t & 31;
    int tile = t >> 5;
    int nt = tile % nNT, kt = tile / nNT;
    int half = lane >> 4, lr = lane & 15;
    int n = nt * 16 + lr;
    int kbase = kt * 32 + half * 8;
    _Float16* dst = Bpk + (size_t)t * 16;
    for (int e = 0; e < 16; ++e) {
        int k = kbase + (e < 8 ? e : e + 8);   // elems 0..7 -> kbase+0..7, 8..15 -> kbase+16..23
        float v = (k < Kreal && n < Nreal) ? W[k * Nreal + n] : 0.0f;
        dst[e] = (_Float16)v;
    }
}

// ---------------------------------------------------------------- WMMA MLP layer
// Y[natoms][NREAL] = act( X[natoms][K] @ Wpacked + bias ). One wave = 16-row tile.
// K in {32,64} (compile time, rows fully dense/padded), NREAL in {64, 8}.
template <int K, int NREAL, bool ACT>
__global__ void mlp_wmma_kernel(const float* __restrict__ X,
                                const _Float16* __restrict__ Bpk,
                                const float* __restrict__ Bias,
                                float* __restrict__ Y) {
    constexpr int NKT  = K / 32;
    constexpr int NPAD = (NREAL + 15) & ~15;
    constexpr int NNT  = NPAD / 16;
    const int NTILES = NATOMS / 16;
    int wave = (blockIdx.x * blockDim.x + threadIdx.x) >> 5;
    if (wave >= NTILES) return;
    int lane = threadIdx.x & 31;
    int half = lane >> 4, lr = lane & 15;
    const float* xr = X + (size_t)(wave * 16 + lr) * K;

    // A fragments: lane holds row (wave*16+lr); halves 0..7 = K[kb..kb+7], 8..15 = K[kb+16..kb+23]
    v16h a[NKT];
#pragma unroll
    for (int kt = 0; kt < NKT; ++kt) {
        int kb = kt * 32 + half * 8;
        float4 f0 = *(const float4*)(xr + kb);
        float4 f1 = *(const float4*)(xr + kb + 4);
        float4 f2 = *(const float4*)(xr + kb + 16);
        float4 f3 = *(const float4*)(xr + kb + 20);
        _Float16* ap = (_Float16*)&a[kt];
        ap[0]  = (_Float16)f0.x; ap[1]  = (_Float16)f0.y;
        ap[2]  = (_Float16)f0.z; ap[3]  = (_Float16)f0.w;
        ap[4]  = (_Float16)f1.x; ap[5]  = (_Float16)f1.y;
        ap[6]  = (_Float16)f1.z; ap[7]  = (_Float16)f1.w;
        ap[8]  = (_Float16)f2.x; ap[9]  = (_Float16)f2.y;
        ap[10] = (_Float16)f2.z; ap[11] = (_Float16)f2.w;
        ap[12] = (_Float16)f3.x; ap[13] = (_Float16)f3.y;
        ap[14] = (_Float16)f3.z; ap[15] = (_Float16)f3.w;
    }

#pragma unroll
    for (int nt = 0; nt < NNT; ++nt) {
        v8f acc = {};
#pragma unroll
        for (int kt = 0; kt < NKT; ++kt) {
            v16h b = *(const v16h*)(Bpk + (size_t)((kt * NNT + nt) * 32 + lane) * 16);
            acc = __builtin_amdgcn_wmma_f32_16x16x32_f16(
                false, a[kt], false, b, (short)0, acc, false, false);
        }
        int n = nt * 16 + lr;
        if ((NREAL % 16 == 0) || (n < NREAL)) {
            float bias = Bias[n];
            float* accp = (float*)&acc;
#pragma unroll
            for (int r = 0; r < 8; ++r) {
                int m = wave * 16 + r + half * 8;   // D layout: lanes 0-15 M=r, 16-31 M=r+8
                float xv = accp[r] + bias;
                if (ACT) xv = xv / (1.0f + __expf(-xv));
                Y[(size_t)m * NREAL + n] = xv;
            }
        }
    }
}

// ---------------------------------------------------------------- edge scatter
// 8 threads per edge; lane owns one wave-channel w. A wave covers 4 consecutive
// edges, so each atomic instruction touches 4 atoms x 32B contiguous.
__global__ void density_edge_kernel(const float* __restrict__ cart,
                                    const int* __restrict__ atomindex,
                                    const float* __restrict__ shifts,
                                    const float* __restrict__ alpha,
                                    const float* __restrict__ rs,
                                    const float* __restrict__ coeffs,
                                    float* __restrict__ mp_sph) {
    int t = blockIdx.x * blockDim.x + threadIdx.x;
    int w = t & 7;
    int e = t >> 3;
    if (e >= NEDGES) return;
    int i = atomindex[e];
    int j = atomindex[NEDGES + e];
    float cx = cart[j]              - cart[i]              + shifts[e];
    float cy = cart[NATOMS + j]     - cart[NATOMS + i]     + shifts[NEDGES + e];
    float cz = cart[2 * NATOMS + j] - cart[2 * NATOMS + i] + shifts[2 * NEDGES + e];
    float d  = sqrtf(cx * cx + cy * cy + cz * cz);
    float dm = fminf(d, CUTOFFF);
    float cv = __cosf(PI_F * dm / CUTOFFF);
    float fc = 0.25f * (cv + 1.0f) * (cv + 1.0f);

    float tw  = d - rs[w];
    float rwv = __expf(-alpha[w] * tw * tw) * fc * coeffs[j * NWAVE + w];

    float x = cx * (1.0f / CUTOFFF), y = cy * (1.0f / CUTOFFF), z = cz * (1.0f / CUTOFFF);
    float r2 = x * x + y * y + z * z;
    float sph[NSPH];
    sph[0] = 0.28209479177387814f;
    sph[1] = 0.4886025119029199f * y;
    sph[2] = 0.4886025119029199f * z;
    sph[3] = 0.4886025119029199f * x;
    sph[4] = 1.0925484305920792f * x * y;
    sph[5] = 1.0925484305920792f * y * z;
    sph[6] = 0.31539156525252005f * (3.0f * z * z - r2);
    sph[7] = 1.0925484305920792f * x * z;
    sph[8] = 0.5462742152960396f * (x * x - y * y);

    float* base = mp_sph + (size_t)i * (NSPH * NWAVE) + w;   // L2-resident target
#pragma unroll
    for (int m = 0; m < NSPH; ++m)
        atomic_add_f32(base + m * NWAVE, sph[m] * rwv);
}

// ---------------------------------------------------------------- dens = sum_l mp_sph^2
__global__ void dens_kernel(const float* __restrict__ mp_sph, float* __restrict__ dens) {
    int t = blockIdx.x * blockDim.x + threadIdx.x;
    if (t >= NATOMS * NORBIT) return;
    int a = t / NORBIT;
    int lw = t - a * NORBIT;
    int l = lw >> 3;          // l-major: orbit = l*NWAVE + w
    int w = lw & 7;
    int m0 = l * l, cnt = 2 * l + 1;
    float s = 0.0f;
    for (int m = m0; m < m0 + cnt; ++m) {
        float v = mp_sph[(size_t)a * 72 + m * NWAVE + w];
        s += v * v;
    }
    dens[(size_t)a * DENS_LD + lw] = s;   // pad cols 24..31 stay zero
}

// ---------------------------------------------------------------- final 64->1 + sum
__global__ void out_final_kernel(const float* __restrict__ h,
                                 const float* __restrict__ w,
                                 const float* __restrict__ b,
                                 float* __restrict__ out) {
    __shared__ float red[256];
    int a = blockIdx.x * blockDim.x + threadIdx.x;
    float s = 0.0f;
    if (a < NATOMS) {
        const float* hp = h + (size_t)a * 64;
#pragma unroll
        for (int k = 0; k < 64; ++k) s += hp[k] * w[k];
        s += b[0];
    }
    red[threadIdx.x] = s;
    __syncthreads();
    for (int off = 128; off > 0; off >>= 1) {
        if (threadIdx.x < off) red[threadIdx.x] += red[threadIdx.x + off];
        __syncthreads();
    }
    if (threadIdx.x == 0) atomic_add_f32(out, red[0]);
}

// ================================================================ host side
extern "C" void kernel_launch(void* const* d_in, const int* in_sizes, int n_in,
                              void* d_out, int out_size, void* d_ws, size_t ws_size,
                              hipStream_t stream) {
    const float* cart      = (const float*)d_in[0];
    const int*   atomindex = (const int*)  d_in[1];
    const float* shifts    = (const float*)d_in[2];
    const float* species   = (const float*)d_in[3];
    const float* alpha     = (const float*)d_in[4];
    const float* rs        = (const float*)d_in[5];
    const float* emb_w[3] = {(const float*)d_in[6],  (const float*)d_in[8],  (const float*)d_in[10]};
    const float* emb_b[3] = {(const float*)d_in[7],  (const float*)d_in[9],  (const float*)d_in[11]};
    const float* mp_w[2][3] = {
        {(const float*)d_in[12], (const float*)d_in[14], (const float*)d_in[16]},
        {(const float*)d_in[18], (const float*)d_in[20], (const float*)d_in[22]}};
    const float* mp_b[2][3] = {
        {(const float*)d_in[13], (const float*)d_in[15], (const float*)d_in[17]},
        {(const float*)d_in[19], (const float*)d_in[21], (const float*)d_in[23]}};
    const float* out_w[3] = {(const float*)d_in[24], (const float*)d_in[26], (const float*)d_in[28]};
    const float* out_b[3] = {(const float*)d_in[25], (const float*)d_in[27], (const float*)d_in[29]};

    // ---- workspace carve-up (f32 region, then f16 packed-weight region) ----
    float* ws     = (float*)d_ws;
    float* mp_sph = ws;  ws += NATOMS * NSPH * NWAVE;   // 1,440,000
    float* coeffs = ws;  ws += NATOMS * NWAVE;          //   160,000
    float* dens   = ws;  ws += NATOMS * DENS_LD;        //   640,000
    float* h1     = ws;  ws += NATOMS * 64;             // 1,280,000
    float* h2     = ws;  ws += NATOMS * 64;             // 1,280,000
    _Float16* fp = (_Float16*)ws;                       // 32B-aligned (all counts mult. of 8)
    // packed weights: [emb_w1][emb_w2][mp0_w0][mp0_w1][mp0_w2][mp1_w0][mp1_w1][mp1_w2][out_w0][out_w1]
    const int SZ_K64N64 = 2 * 4 * 32 * 16;   // 4096 halves
    const int SZ_K64N16 = 2 * 1 * 32 * 16;   // 1024 halves
    const int SZ_K32N64 = 1 * 4 * 32 * 16;   // 2048 halves
    _Float16* pk_emb_w1 = fp;                 fp += SZ_K64N64;
    _Float16* pk_emb_w2 = fp;                 fp += SZ_K64N16;
    _Float16* pk_mp_w0[2]; _Float16* pk_mp_w1[2]; _Float16* pk_mp_w2[2];
    for (int p = 0; p < 2; ++p) {
        pk_mp_w0[p] = fp; fp += SZ_K32N64;
        pk_mp_w1[p] = fp; fp += SZ_K64N64;
        pk_mp_w2[p] = fp; fp += SZ_K64N16;
    }
    _Float16* pk_out_w0 = fp;                 fp += SZ_K32N64;
    _Float16* pk_out_w1 = fp;                 fp += SZ_K64N64;

    const int TPB = 256;
    const int GEMM_BLOCKS = (NATOMS / 16 * 32 + TPB - 1) / TPB;        // 157
    const int EDGE_BLOCKS = (NEDGES * NWAVE + TPB - 1) / TPB;          // 31250

    // ---- zero persistent accumulator, padded dens, output ----
    zero_kernel<<<512, TPB, 0, stream>>>(mp_sph, NATOMS * NSPH * NWAVE);
    zero_kernel<<<512, TPB, 0, stream>>>(dens, NATOMS * DENS_LD);
    zero_kernel<<<1, 32, 0, stream>>>((float*)d_out, out_size);

    // ---- pre-pack all GEMM weights into WMMA B layout ----
    pack_w_kernel<<<1, 256, 0, stream>>>(emb_w[1], 64, 64, 2, 4, pk_emb_w1);
    pack_w_kernel<<<1, 64,  0, stream>>>(emb_w[2], 64,  8, 2, 1, pk_emb_w2);
    for (int p = 0; p < 2; ++p) {
        pack_w_kernel<<<1, 128, 0, stream>>>(mp_w[p][0], 24, 64, 1, 4, pk_mp_w0[p]);
        pack_w_kernel<<<1, 256, 0, stream>>>(mp_w[p][1], 64, 64, 2, 4, pk_mp_w1[p]);
        pack_w_kernel<<<1, 64,  0, stream>>>(mp_w[p][2], 64,  8, 2, 1, pk_mp_w2[p]);
    }
    pack_w_kernel<<<1, 128, 0, stream>>>(out_w[0], 24, 64, 1, 4, pk_out_w0);
    pack_w_kernel<<<1, 256, 0, stream>>>(out_w[1], 64, 64, 2, 4, pk_out_w1);

    // ---- embedding MLP: 1 -> 64 -> 64 -> 8 ----
    emb_l1_kernel<<<(NATOMS * 64 + TPB - 1) / TPB, TPB, 0, stream>>>(species, emb_w[0], emb_b[0], h1);
    mlp_wmma_kernel<64, 64, true ><<<GEMM_BLOCKS, TPB, 0, stream>>>(h1, pk_emb_w1, emb_b[1], h2);
    mlp_wmma_kernel<64,  8, false><<<GEMM_BLOCKS, TPB, 0, stream>>>(h2, pk_emb_w2, emb_b[2], coeffs);

    // ---- two message-passing iterations ----
    for (int p = 0; p < 2; ++p) {
        density_edge_kernel<<<EDGE_BLOCKS, TPB, 0, stream>>>(cart, atomindex, shifts, alpha, rs, coeffs, mp_sph);
        dens_kernel<<<(NATOMS * NORBIT + TPB - 1) / TPB, TPB, 0, stream>>>(mp_sph, dens);
        mlp_wmma_kernel<32, 64, true ><<<GEMM_BLOCKS, TPB, 0, stream>>>(dens, pk_mp_w0[p], mp_b[p][0], h1);
        mlp_wmma_kernel<64, 64, true ><<<GEMM_BLOCKS, TPB, 0, stream>>>(h1, pk_mp_w1[p], mp_b[p][1], h2);
        mlp_wmma_kernel<64,  8, false><<<GEMM_BLOCKS, TPB, 0, stream>>>(h2, pk_mp_w2[p], mp_b[p][2], coeffs);
    }

    // ---- final density + readout MLP: 24 -> 64 -> 64 -> 1, summed over atoms ----
    density_edge_kernel<<<EDGE_BLOCKS, TPB, 0, stream>>>(cart, atomindex, shifts, alpha, rs, coeffs, mp_sph);
    dens_kernel<<<(NATOMS * NORBIT + TPB - 1) / TPB, TPB, 0, stream>>>(mp_sph, dens);
    mlp_wmma_kernel<32, 64, true ><<<GEMM_BLOCKS, TPB, 0, stream>>>(dens, pk_out_w0, out_b[0], h1);
    mlp_wmma_kernel<64, 64, true ><<<GEMM_BLOCKS, TPB, 0, stream>>>(h1, pk_out_w1, out_b[1], h2);
    out_final_kernel<<<(NATOMS + TPB - 1) / TPB, TPB, 0, stream>>>(h2, out_w[2], out_b[2], (float*)d_out);

    (void)in_sizes; (void)n_in; (void)ws_size; (void)fp;
}
